// AFMA_27273042330243
// MI455X (gfx1250) — matrix operand order repaired and checked by reading, not compiled, not run on hardware
//
#include <hip/hip_runtime.h>
#include <hip/hip_bf16.h>
#include <stdint.h>

// CDNA5 / gfx1250: wave32, WMMA 16x16x32 bf16, 320KB LDS per WGP.

typedef __attribute__((ext_vector_type(16))) __bf16 v16bf;
typedef __attribute__((ext_vector_type(8)))  __bf16 v8bf;
typedef __attribute__((ext_vector_type(8)))  float  v8f;
typedef __attribute__((ext_vector_type(4)))  float  f4;

#define B_   4
#define C_   32
#define H_   320
#define W_   320
#define HW_  (H_ * W_)      // 102400
#define N_   1024           // patches per image (32x32)
#define Q_   100            // pixels per patch (10x10)
#define QP_  112            // Q padded to 7 WMMA col-tiles of 16
#define LSTR 1048           // LDS row stride in bf16 elems (bank-conflict-free pad)

// ---------------------------------------------------------------------------
// k0: per-pixel argmax over channels -> transposed one-hot index map
//     amapT[b][q][m] = argmax_c x[b,c,pixel(m,q)]   (uint8)
// ---------------------------------------------------------------------------
__global__ void afma_argmax_kernel(const float* __restrict__ x,
                                   uint8_t* __restrict__ amapT) {
    int p = blockIdx.x * blockDim.x + threadIdx.x;
    if (p >= B_ * HW_) return;
    int b   = p / HW_;
    int rem = p - b * HW_;
    int h   = rem / W_;
    int w   = rem - h * W_;
    const float* xp = x + (size_t)b * C_ * HW_ + rem;
    float best = xp[0];
    int   bi   = 0;
#pragma unroll
    for (int c = 1; c < C_; ++c) {
        float v = xp[(size_t)c * HW_];
        if (v > best) { best = v; bi = c; }   // strict > keeps first index (jnp.argmax)
    }
    int m = (h / 10) * 32 + (w / 10);
    int q = (h % 10) * 10 + (w % 10);
    amapT[(size_t)b * (Q_ * N_) + q * N_ + m] = (uint8_t)bi;
}

// ---------------------------------------------------------------------------
// k1: fused passthrough-copy of attentions (2nd output, non-temporal: zero
//     reuse, keep it out of L2) + per-row nonzero count
//     -> invnz[row] = 1/(count + 1e-5).  One wave32 per 1024-elem row.
// ---------------------------------------------------------------------------
__global__ void afma_nz_copy_kernel(const float* __restrict__ attn,
                                    float* __restrict__ attn_out,
                                    float* __restrict__ invnz) {
    int wave = threadIdx.x >> 5;
    int lane = threadIdx.x & 31;
    int row  = blockIdx.x * 8 + wave;          // B_*C_*N_ = 131072 rows total
    const f4* src = (const f4*)(attn     + (size_t)row * N_);
    f4*       dst = (f4*)      (attn_out + (size_t)row * N_);
    int cnt = 0;
#pragma unroll
    for (int j = 0; j < 8; ++j) {
        int idx = j * 32 + lane;               // 256 float4 per row
        f4 v = __builtin_nontemporal_load(src + idx);
        __builtin_nontemporal_store(v, dst + idx);
        cnt += (v.x != 0.0f) + (v.y != 0.0f) + (v.z != 0.0f) + (v.w != 0.0f);
    }
#pragma unroll
    for (int off = 16; off; off >>= 1) cnt += __shfl_down(cnt, off, 32);
    if (lane == 0) invnz[row] = 1.0f / ((float)cnt + 1e-5f);
}

// ---------------------------------------------------------------------------
// helper: load one A fragment (16x32 A operand in this code's contiguous K order) from a global fp32
// row and convert to bf16 in registers.
//   lane<16 holds K = ks+{0..15}; lane>=16 holds K = ks+{16..31}
// ---------------------------------------------------------------------------
__device__ __forceinline__ v16bf load_a_frag(const float* __restrict__ ap) {
    f4 f0 = *(const f4*)(ap + 0);
    f4 f1 = *(const f4*)(ap + 4);
    f4 f2 = *(const f4*)(ap + 8);
    f4 f3 = *(const f4*)(ap + 12);
    v16bf a;
    a[0]  = (__bf16)f0.x; a[1]  = (__bf16)f0.y; a[2]  = (__bf16)f0.z; a[3]  = (__bf16)f0.w;
    a[4]  = (__bf16)f1.x; a[5]  = (__bf16)f1.y; a[6]  = (__bf16)f1.z; a[7]  = (__bf16)f1.w;
    a[8]  = (__bf16)f2.x; a[9]  = (__bf16)f2.y; a[10] = (__bf16)f2.z; a[11] = (__bf16)f2.w;
    a[12] = (__bf16)f3.x; a[13] = (__bf16)f3.y; a[14] = (__bf16)f3.z; a[15] = (__bf16)f3.w;
    return a;
}

// helper: load one B fragment (32x16, 16 contiguous bf16 per lane) from LDS
__device__ __forceinline__ v16bf load_b_frag(const __bf16* bp) {
    union { v16bf v; v8bf h[2]; } bu;
    bu.h[0] = *(const v8bf*)(bp);
    bu.h[1] = *(const v8bf*)(bp + 8);
    return bu.v;
}

// ---------------------------------------------------------------------------
// k2: per-(b,c) GEMM  att[1024x100(->112)] = A[1024x1024] * u[1024x100]
//     u (one-hot/4, bf16) staged transposed in LDS once per workgroup.
//     Each wave owns pairs of row tiles (M-blocking x2).  B fragments use a
//     rotating double buffer: live regs stay low (no scratch spills) while
//     each WMMA's B registers are only recycled 4 WMMAs later, matching the
//     bf16 WMMA WAR hazard window (4 co-exec slots) without v_nops.
//     One 235KB-LDS workgroup per WGP -> __launch_bounds__(256,1) frees the
//     full VGPR budget (14 f32 accumulators = 112 VGPRs live).
//     Epilogue: scale rows by invnz, fold to image, out = att*x + x.
// ---------------------------------------------------------------------------
__global__ __launch_bounds__(256, 1)
void afma_gemm_kernel(const float* __restrict__ attn,
                      const float* __restrict__ x,
                      const uint8_t* __restrict__ amapT,
                      const float* __restrict__ invnz,
                      float* __restrict__ out) {
    extern __shared__ __bf16 lds[];            // [QP_][LSTR] transposed u
    const int bc  = blockIdx.x;                // 0..127 == (b,c)
    const int b   = bc >> 5;
    const int c   = bc & 31;
    const int tid = threadIdx.x;

    // ---- stage u^T into LDS (coalesced amapT reads along m) ----
    const uint8_t* am = amapT + (size_t)b * (Q_ * N_);
    for (int idx = tid; idx < QP_ * N_; idx += 256) {
        int q = idx >> 10;                     // 0..111
        int m = idx & (N_ - 1);                // 0..1023
        __bf16 v = (__bf16)0.0f;
        if (q < Q_ && am[q * N_ + m] == (uint8_t)c) v = (__bf16)0.25f;
        lds[q * LSTR + m] = v;
    }
    __syncthreads();

    const int wave  = tid >> 5;
    const int lane  = tid & 31;
    const int lhalf = (lane >> 4) & 1;         // 0: lanes 0-15, 1: lanes 16-31
    const int l15   = lane & 15;
    const float* Abc   = attn  + (size_t)bc * N_ * N_;
    const float* invbc = invnz + (size_t)bc * N_;

    // per-lane LDS base for B fragments: column q = t*16+l15, K-half by lhalf
    const __bf16* bbase = lds + (size_t)l15 * LSTR + (lhalf ? 16 : 0);

    for (int pr = wave; pr < 32; pr += 8) {    // 32 row-tile pairs, uniform/wave
        const int rt0 = pr * 2, rt1 = pr * 2 + 1;
        const float* arow0 = Abc + (size_t)(rt0 * 16 + l15) * N_ + (lhalf ? 16 : 0);
        const float* arow1 = Abc + (size_t)(rt1 * 16 + l15) * N_ + (lhalf ? 16 : 0);

        v8f acc0[7], acc1[7];
#pragma unroll
        for (int t = 0; t < 7; ++t) {
            acc0[t] = (v8f){0,0,0,0,0,0,0,0};
            acc1[t] = (v8f){0,0,0,0,0,0,0,0};
        }

        for (int kb = 0; kb < 32; ++kb) {      // K = 1024 in steps of 32
            const int ks = kb * 32;
            __builtin_prefetch(arow0 + ks + 256, 0, 1);  // global_prefetch next A
            __builtin_prefetch(arow1 + ks + 256, 0, 1);

            // A fragments for both row tiles (fp32 -> bf16 in regs)
            v16bf a0 = load_a_frag(arow0 + ks);
            v16bf a1 = load_a_frag(arow1 + ks);

            // rotating double-buffered B; 14 WMMAs per K-step
            v16bf bcur = load_b_frag(bbase + ks);        // t = 0
#pragma unroll
            for (int t = 0; t < 7; ++t) {
                v16bf bnext = bcur;
                if (t < 6)
                    bnext = load_b_frag(bbase + (size_t)(t + 1) * 16 * LSTR + ks);
                acc0[t] = __builtin_amdgcn_wmma_f32_16x16x32_bf16(
                    false, a0, false, bcur, (short)0, acc0[t], false, false);
                acc1[t] = __builtin_amdgcn_wmma_f32_16x16x32_bf16(
                    false, a1, false, bcur, (short)0, acc1[t], false, false);
                bcur = bnext;
            }
        }

        // ---- epilogue: scale by 1/nz, fold, out = att*x + x ----
#pragma unroll
        for (int half = 0; half < 2; ++half) {
            const int rt = half ? rt1 : rt0;
            const v8f* acc = half ? acc1 : acc0;
            float inv_r[8];
#pragma unroll
            for (int r = 0; r < 8; ++r)
                inv_r[r] = invbc[rt * 16 + r + lhalf * 8];
#pragma unroll
            for (int t = 0; t < 7; ++t) {
                int q = t * 16 + l15;          // output column (pixel in patch)
                if (q >= Q_) continue;         // padding columns
                int qh = q / 10, qw = q - qh * 10;
#pragma unroll
                for (int r = 0; r < 8; ++r) {
                    int n  = rt * 16 + r + lhalf * 8;   // output row (patch idx)
                    int ph = n >> 5, pw = n & 31;
                    size_t xi = (size_t)bc * HW_
                              + (size_t)(ph * 10 + qh) * W_ + (pw * 10 + qw);
                    float xv = x[xi];
                    float v  = acc[t][r] * inv_r[r];
                    out[xi]  = v * xv + xv;
                }
            }
        }
    }
}

// ---------------------------------------------------------------------------
extern "C" void kernel_launch(void* const* d_in, const int* in_sizes, int n_in,
                              void* d_out, int out_size, void* d_ws, size_t ws_size,
                              hipStream_t stream) {
    const float* x    = (const float*)d_in[0];
    const float* attn = (const float*)d_in[1];
    float* out      = (float*)d_out;
    float* attn_out = out + (size_t)B_ * C_ * HW_;          // 2nd tuple output

    uint8_t* amapT = (uint8_t*)d_ws;                        // 409,600 B
    float*   invnz = (float*)((uint8_t*)d_ws + 524288);     // 524,288 B

    afma_argmax_kernel<<<(B_ * HW_ + 255) / 256, 256, 0, stream>>>(x, amapT);
    afma_nz_copy_kernel<<<(B_ * C_ * N_) / 8, 256, 0, stream>>>(attn, attn_out, invnz);

    const size_t ldsBytes = (size_t)QP_ * LSTR * sizeof(__bf16);  // 234,752 B (<320KB/WGP)
    (void)hipFuncSetAttribute(reinterpret_cast<const void*>(afma_gemm_kernel),
                              hipFuncAttributeMaxDynamicSharedMemorySize,
                              (int)ldsBytes);
    afma_gemm_kernel<<<B_ * C_, 256, ldsBytes, stream>>>(attn, x, amapT, invnz, out);
}
